// CrossAssetAttention_49108656063281
// MI455X (gfx1250) — compile-verified, matrix operand order checked by reading, not run on hardware
//
#include <hip/hip_runtime.h>
#include <hip/hip_bf16.h>

typedef _Float16 f16;
typedef _Float16 v16h __attribute__((ext_vector_type(16)));
typedef _Float16 v8h  __attribute__((ext_vector_type(8)));
typedef _Float16 v4h  __attribute__((ext_vector_type(4)));
typedef float    v8f  __attribute__((ext_vector_type(8)));
typedef float    v4f  __attribute__((ext_vector_type(4)));

#define HS    256   // hidden
#define SEQ   64    // sequence length
#define NHEAD 4
#define HDIM  64

// ---- LDS layout (bytes). Phase overlay: YF reuses QH..AACC space. ----
#define XH_OFF   0         // f16 [64][264]  x (converted)
#define XH_STR   264
#define QH_OFF   33792     // f16 [64][72]
#define KH_OFF   43008     // f16 [64][72]
#define VT_OFF   52224     // f16 [64][72]  V transposed: VT[d][kk]
#define HSTR     72
#define SF_OFF   61440     // f32 [64][68]  raw scores
#define SF_STR   68
#define SH_OFF   78848     // f16 [64][72]  softmax probs
#define AH_OFF   88064     // f16 [64][72]  attended (per head)
#define AACC_OFF 97280     // f32 [64][64]  attn average accumulator
#define YF_OFF   33792     // f32 [64][264] residual+out (overlay, epilogue only)
#define YF_STR   264
#define WS_OFF   113664    // weight staging: B1 view f16[64][264], B5 view f16[256][72]
#define WSB_STR  264
#define WS5_STR  72
#define SMEM_BYTES 150528  // 147 KB -> 2 workgroups / WGP (<=320 KB)

// ---- async global->LDS staging (guarded; falls back to plain copies) ----
#if defined(__has_builtin)
#if __has_builtin(__builtin_amdgcn_global_load_async_to_lds_b128) && \
    __has_builtin(__builtin_amdgcn_s_wait_asynccnt)
#define USE_ASYNC_LDS 1
#endif
#endif

#define AS1 __attribute__((address_space(1)))
#define AS3 __attribute__((address_space(3)))
typedef int gvec4i __attribute__((vector_size(16)));   // matches builtin's V4i

__device__ __forceinline__ void g2l_16B(f16* lptr, const f16* gptr) {
#ifdef USE_ASYNC_LDS
  __builtin_amdgcn_global_load_async_to_lds_b128(
      (AS1 gvec4i*)(uintptr_t)gptr,
      (AS3 gvec4i*)(unsigned int)(uintptr_t)lptr, 0, 0);
#else
  *(v8h*)lptr = *(const v8h*)gptr;
#endif
}
__device__ __forceinline__ void g2l_fence() {
#ifdef USE_ASYNC_LDS
  __builtin_amdgcn_s_wait_asynccnt(0);
#endif
}

// Assemble a 16x32 (A) or 32x16 (B) f16 WMMA fragment: per-lane two
// contiguous 16B chunks (K = k0..k0+7 and k0+16..k0+23).
__device__ __forceinline__ v16h load_frag(const f16* p0, const f16* p1) {
  v8h lo = *(const v8h*)p0;
  v8h hi = *(const v8h*)p1;
  v16h r;
#pragma unroll
  for (int i = 0; i < 8; ++i) { r[i] = lo[i]; r[i + 8] = hi[i]; }
  return r;
}

__global__ void wconv_kernel(const float* __restrict__ Wq, const float* __restrict__ Wk,
                             const float* __restrict__ Wv, const float* __restrict__ Wo,
                             f16* __restrict__ o) {
  int i = blockIdx.x * 256 + threadIdx.x;           // 65536 threads total
  o[i]           = (f16)Wq[i];
  o[i +  65536]  = (f16)Wk[i];
  o[i + 131072]  = (f16)Wv[i];
  o[i + 196608]  = (f16)Wo[i];
}

__global__ __launch_bounds__(256)
void attn_kernel(const float* __restrict__ x, const f16* __restrict__ Wh,
                 const float* __restrict__ bq, const float* __restrict__ bk,
                 const float* __restrict__ bv, const float* __restrict__ bo,
                 const float* __restrict__ g,  const float* __restrict__ bb,
                 float* __restrict__ out, float* __restrict__ outAttn) {
  __shared__ __align__(16) unsigned char smem[SMEM_BYTES];
  f16*   XH   = (f16*)(smem + XH_OFF);
  f16*   QH   = (f16*)(smem + QH_OFF);
  f16*   KH   = (f16*)(smem + KH_OFF);
  f16*   VT   = (f16*)(smem + VT_OFF);
  float* SF   = (float*)(smem + SF_OFF);
  f16*   SH   = (f16*)(smem + SH_OFF);
  f16*   AH   = (f16*)(smem + AH_OFF);
  float* AACC = (float*)(smem + AACC_OFF);
  float* YF   = (float*)(smem + YF_OFF);
  f16*   WSB  = (f16*)(smem + WS_OFF);   // 64 x 264 view (Q/K/V slice)
  f16*   WS5  = (f16*)(smem + WS_OFF);   // 256 x 72 view (Wo slice)

  const int tid = threadIdx.x;
  const int w   = tid >> 5;      // wave id 0..7
  const int ln  = tid & 31;      // lane
  const int l15 = ln & 15;
  const int hl  = ln >> 4;       // half-wave select
  const int kb  = hl * 8;        // K sub-base per fragment layout
  const int b   = blockIdx.x;
  const float* xb = x + (size_t)b * (SEQ * HS);

  // Warm L2 with the weight lines this block is about to stream.
  __builtin_prefetch(Wh + (size_t)tid * 128, 0, 3);           // Q
  __builtin_prefetch(Wh + 196608 + (size_t)tid * 128, 0, 3);  // O

  // ---- load x -> LDS as f16 (lane-consecutive float4 => coalesced) ----
  for (int c = tid; c < SEQ * HS / 4; c += 256) {
    v4f v = *(const v4f*)(xb + c * 4);
    int row = c >> 6;            // 64 float4 per row
    int col = (c & 63) * 4;
    v4h hv; hv[0] = (f16)v.x; hv[1] = (f16)v.y; hv[2] = (f16)v.z; hv[3] = (f16)v.w;
    *(v4h*)(XH + row * XH_STR + col) = hv;
  }
  __syncthreads();

  v8f acc[8];                        // out-projection C tiles, persist over heads
  const v8f vzero = {0.f,0.f,0.f,0.f,0.f,0.f,0.f,0.f};
#pragma unroll
  for (int i = 0; i < 8; ++i) acc[i] = vzero;

  for (int h = 0; h < NHEAD; ++h) {
    // ---- B1: Q,K,V projections, one matrix at a time with LDS-staged weights ----
    for (int mat = 0; mat < 3; ++mat) {
      const f16* Wslice = Wh + mat * 65536 + h * HDIM * HS;  // 64 rows x 256 cols
      // stage 64x256 f16 slice -> WSB (2048 16B chunks, 8 per thread)
#pragma unroll
      for (int i = 0; i < 8; ++i) {
        int c = tid + i * 256;
        int row = c >> 5, q = c & 31;           // 32 chunks per row
        g2l_16B(WSB + row * WSB_STR + q * 8, Wslice + row * HS + q * 8);
      }
      g2l_fence();
      __syncthreads();

      const float* bias = (mat == 0) ? bq : ((mat == 1) ? bk : bv);
      const int nt  = w & 3;
      const int mtb = (w >> 2) * 2;              // two m-tiles per wave
      const int jl  = nt * 16 + l15;
      const int jg  = h * HDIM + jl;
      v8f c0 = vzero, c1 = vzero;
#pragma unroll
      for (int kk = 0; kk < 8; ++kk) {
        int k0 = kk * 32 + kb;
        v16h bf = load_frag(WSB + jl * WSB_STR + k0, WSB + jl * WSB_STR + k0 + 16);
        v16h a0 = load_frag(XH + (mtb * 16 + l15) * XH_STR + k0,
                            XH + (mtb * 16 + l15) * XH_STR + k0 + 16);
        v16h a1 = load_frag(XH + ((mtb + 1) * 16 + l15) * XH_STR + k0,
                            XH + ((mtb + 1) * 16 + l15) * XH_STR + k0 + 16);
        c0 = __builtin_amdgcn_wmma_f32_16x16x32_f16(false, a0, false, bf, (short)0, c0, false, false);
        c1 = __builtin_amdgcn_wmma_f32_16x16x32_f16(false, a1, false, bf, (short)0, c1, false, false);
      }
      float bs = bias[jg];
#pragma unroll
      for (int mi = 0; mi < 2; ++mi) {
        int mt = mtb + mi;
        v8f c = mi ? c1 : c0;
        if (mat == 2) {                 // V stored transposed: VT[d][a]
          v8h pack;
#pragma unroll
          for (int r = 0; r < 8; ++r) pack[r] = (f16)(c[r] + bs);
          *(v8h*)(VT + jl * HSTR + mt * 16 + hl * 8) = pack;
        } else {
          f16* dst = (mat == 0) ? QH : KH;
#pragma unroll
          for (int r = 0; r < 8; ++r) {
            int ar = mt * 16 + hl * 8 + r;
            dst[ar * HSTR + jl] = (f16)(c[r] + bs);
          }
        }
      }
      __syncthreads();   // before next slice overwrites WSB
    }

    // ---- B2: scores = Q K^T / 8 (16 tiles, K=64) ----
#pragma unroll
    for (int ti = 0; ti < 2; ++ti) {
      int t = w + ti * 8;
      int mt = t >> 2, nt = t & 3;
      int qrow = mt * 16 + l15;
      int krow = nt * 16 + l15;
      v8f c = vzero;
#pragma unroll
      for (int kk = 0; kk < 2; ++kk) {
        int k0 = kk * 32 + kb;
        v16h a  = load_frag(QH + qrow * HSTR + k0, QH + qrow * HSTR + k0 + 16);
        v16h bf = load_frag(KH + krow * HSTR + k0, KH + krow * HSTR + k0 + 16);
        c = __builtin_amdgcn_wmma_f32_16x16x32_f16(false, a, false, bf, (short)0, c, false, false);
      }
      int kcol = nt * 16 + l15;
#pragma unroll
      for (int r = 0; r < 8; ++r) {
        int qr = mt * 16 + hl * 8 + r;
        SF[qr * SF_STR + kcol] = c[r] * 0.125f;   // 1/sqrt(64)
      }
    }
    __syncthreads();

    // ---- B3: row softmax (wave32 shuffle reductions) + attn-avg accumulate ----
    for (int row = w; row < SEQ; row += 8) {
      float e0 = SF[row * SF_STR + ln];
      float e1 = SF[row * SF_STR + ln + 32];
      float m = fmaxf(e0, e1);
#pragma unroll
      for (int off = 16; off > 0; off >>= 1) m = fmaxf(m, __shfl_xor(m, off, 32));
      e0 = __expf(e0 - m);
      e1 = __expf(e1 - m);
      float s = e0 + e1;
#pragma unroll
      for (int off = 16; off > 0; off >>= 1) s += __shfl_xor(s, off, 32);
      float inv = 1.0f / s;
      float p0 = e0 * inv, p1 = e1 * inv;
      SH[row * HSTR + ln]      = (f16)p0;
      SH[row * HSTR + ln + 32] = (f16)p1;
      if (h == 0) {
        AACC[row * 64 + ln]      = p0;
        AACC[row * 64 + ln + 32] = p1;
      } else {
        AACC[row * 64 + ln]      += p0;
        AACC[row * 64 + ln + 32] += p1;
      }
    }
    __syncthreads();

    // ---- B4: attended = P @ V (16 tiles, K=64; B loads from VT contiguous) ----
#pragma unroll
    for (int ti = 0; ti < 2; ++ti) {
      int t = w + ti * 8;
      int mt = t >> 2, nt = t & 3;
      int qrow = mt * 16 + l15;
      int dcol = nt * 16 + l15;
      v8f c = vzero;
#pragma unroll
      for (int kk = 0; kk < 2; ++kk) {
        int k0 = kk * 32 + kb;
        v16h a  = load_frag(SH + qrow * HSTR + k0, SH + qrow * HSTR + k0 + 16);
        v16h bf = load_frag(VT + dcol * HSTR + k0, VT + dcol * HSTR + k0 + 16);
        c = __builtin_amdgcn_wmma_f32_16x16x32_f16(false, a, false, bf, (short)0, c, false, false);
      }
#pragma unroll
      for (int r = 0; r < 8; ++r) {
        int qr = mt * 16 + hl * 8 + r;
        AH[qr * HSTR + dcol] = (f16)c[r];
      }
    }
    __syncthreads();

    // ---- B5: out += attended_h @ Wo_h^T with LDS-staged Wo slice ----
    {
      const f16* Wo_m = Wh + 196608;
      // stage Wo[:, h*64 : h*64+64] -> WS5 [256][72] (2048 16B chunks)
#pragma unroll
      for (int i = 0; i < 8; ++i) {
        int c = tid + i * 256;
        int row = c >> 3, q = c & 7;            // 8 chunks per row
        g2l_16B(WS5 + row * WS5_STR + q * 8, Wo_m + row * HS + h * HDIM + q * 8);
      }
      g2l_fence();
      __syncthreads();

      for (int mt = 0; mt < 4; ++mt) {
#pragma unroll
        for (int ni = 0; ni < 2; ++ni) {
          int nt = 2 * w + ni;
          int arow = mt * 16 + l15;
          int j = nt * 16 + l15;
          v8f c = acc[mt * 2 + ni];
#pragma unroll
          for (int kk = 0; kk < 2; ++kk) {
            int k0 = kk * 32 + kb;
            v16h a  = load_frag(AH + arow * HSTR + k0, AH + arow * HSTR + k0 + 16);
            v16h bf = load_frag(WS5 + j * WS5_STR + k0, WS5 + j * WS5_STR + k0 + 16);
            c = __builtin_amdgcn_wmma_f32_16x16x32_f16(false, a, false, bf, (short)0, c, false, false);
          }
          acc[mt * 2 + ni] = c;
        }
      }
      __syncthreads();   // protect WSB restaging next head
    }
  }

  // ---- epilogue 1: write attn_avg = AACC * 0.25 ----
  {
    float* oa = outAttn + (size_t)b * (SEQ * SEQ);
    int idx = tid * 16;
#pragma unroll
    for (int i = 0; i < 16; i += 4) {
      v4f v;
      v.x = AACC[idx + i + 0] * 0.25f;
      v.y = AACC[idx + i + 1] * 0.25f;
      v.z = AACC[idx + i + 2] * 0.25f;
      v.w = AACC[idx + i + 3] * 0.25f;
      *(v4f*)(oa + idx + i) = v;
    }
  }
  __syncthreads();   // before YF overlays QH..AACC space

  // ---- epilogue 2: y = x + out + bo -> YF (f32) ----
  for (int mt = 0; mt < 4; ++mt) {
#pragma unroll
    for (int ni = 0; ni < 2; ++ni) {
      int nt = 2 * w + ni;
      int j = nt * 16 + l15;
      float bos = bo[j];
      v8f c = acc[mt * 2 + ni];
#pragma unroll
      for (int r = 0; r < 8; ++r) {
        int ar = mt * 16 + hl * 8 + r;
        YF[ar * YF_STR + j] = (float)XH[ar * XH_STR + j] + c[r] + bos;
      }
    }
  }
  __syncthreads();

  // ---- epilogue 3: LayerNorm per row (wave32 reductions), write output ----
  for (int row = w; row < SEQ; row += 8) {
    const float* yr = YF + row * YF_STR;
    float s = 0.f;
#pragma unroll
    for (int i = 0; i < 8; ++i) s += yr[ln + 32 * i];
#pragma unroll
    for (int off = 16; off > 0; off >>= 1) s += __shfl_xor(s, off, 32);
    float mu = s * (1.0f / HS);
    float v = 0.f;
#pragma unroll
    for (int i = 0; i < 8; ++i) { float d = yr[ln + 32 * i] - mu; v += d * d; }
#pragma unroll
    for (int off = 16; off > 0; off >>= 1) v += __shfl_xor(v, off, 32);
    float rstd = rsqrtf(v * (1.0f / HS) + 1e-5f);
    float* orow = out + (size_t)b * (SEQ * HS) + row * HS;
#pragma unroll
    for (int i = 0; i < 8; ++i) {
      int j = ln + 32 * i;
      orow[j] = (yr[j] - mu) * rstd * g[j] + bb[j];
    }
  }
}

extern "C" void kernel_launch(void* const* d_in, const int* in_sizes, int n_in,
                              void* d_out, int out_size, void* d_ws, size_t ws_size,
                              hipStream_t stream) {
  const float* x  = (const float*)d_in[0];
  const float* Wq = (const float*)d_in[1];
  const float* bq = (const float*)d_in[2];
  const float* Wk = (const float*)d_in[3];
  const float* bk = (const float*)d_in[4];
  const float* Wv = (const float*)d_in[5];
  const float* bv = (const float*)d_in[6];
  const float* Wo = (const float*)d_in[7];
  const float* bo = (const float*)d_in[8];
  const float* g  = (const float*)d_in[9];
  const float* bb = (const float*)d_in[10];
  float* out = (float*)d_out;
  f16* Wh = (f16*)d_ws;                       // 4 x 256 x 256 f16 = 512 KB

  int nB = in_sizes[0] / (SEQ * HS);
  float* outAttn = out + (size_t)nB * SEQ * HS;

  wconv_kernel<<<256, 256, 0, stream>>>(Wq, Wk, Wv, Wo, Wh);
  attn_kernel<<<nB, 256, 0, stream>>>(x, Wh, bq, bk, bv, bo, g, bb, out, outAttn);
}
